// CachedDecoder_82514911691465
// MI455X (gfx1250) — compile-verified, matrix operand order checked by reading, not run on hardware
//
#include <hip/hip_runtime.h>
#include <hip/hip_bf16.h>

#define D_MODEL 256
#define SEQ     2048
#define NHEAD   8
#define DKH     32
#define BATCH   4
#define NLAYER  4
#define VOCABN  1000
#define ROWS    (BATCH * SEQ)   // 8192

typedef __attribute__((ext_vector_type(16))) __bf16 v16bf;
typedef __attribute__((ext_vector_type(8)))  __bf16 v8bf;
typedef __attribute__((ext_vector_type(8)))  float  v8f;

// ---------------------------------------------------------------------------
// A/B fragment loader, bf16 source rows (16B-aligned): two 128-bit loads.
// 16-bit A-matrix 16x32 layout (ISA 7.12.2): lane l holds row m = l&15,
// VGPR0..3 -> K = khalf+0..7, VGPR4..7 -> K = 16+khalf+0..7, khalf=8*(l>=16).
// B fragments are rows of B^T (same per-lane layout with N in place of M).
// ---------------------------------------------------------------------------
__device__ __forceinline__ v16bf load_frag_bf16(const __bf16* __restrict__ row,
                                                int khalf) {
  v8bf lo = *(const v8bf*)(row + khalf);
  v8bf hi = *(const v8bf*)(row + 16 + khalf);
  v16bf f;
#pragma unroll
  for (int i = 0; i < 8; ++i) { f[i] = lo[i]; f[i + 8] = hi[i]; }
  return f;
}

__device__ __forceinline__ v8f wmma_bf16(v16bf a, v16bf b, v8f c) {
  return __builtin_amdgcn_wmma_f32_16x16x32_bf16(false, a, false, b,
                                                 (short)0, c, false, false);
}

// ---------------------------------------------------------------------------
// f32 -> bf16 bulk convert (weights, done once per call)
// ---------------------------------------------------------------------------
__global__ void cvt_bf16_kernel(const float* __restrict__ src,
                                __bf16* __restrict__ dst, int n) {
  int i = blockIdx.x * blockDim.x + threadIdx.x;
  if (i < n) dst[i] = (__bf16)src[i];
}

// ---------------------------------------------------------------------------
// Embedding gather: h[row, :] = bf16(embed[x[row], :])
// ---------------------------------------------------------------------------
__global__ void embed_kernel(const int* __restrict__ x,
                             const float* __restrict__ emb,
                             __bf16* __restrict__ h) {
  int row = blockIdx.x;
  int col = threadIdx.x;
  h[(size_t)row * D_MODEL + col] = (__bf16)emb[(size_t)x[row] * D_MODEL + col];
}

// ---------------------------------------------------------------------------
// Y = X @ W^T  (X: [M,256] bf16, W: [N,256] bf16, f32 accumulate)
// MODE 0: Y bf16 row-major [M, N], scaled by oscale   (N == 256)
// MODE 1: Y bf16 per-head-transposed V: vt[((b*H+h)*32+d)*SEQ + s]
// MODE 2: Y f32 = X@W^T + bias, ragged N (vocab head, N = 1000)
// One wave computes a 16x64 tile; 4 waves/block.
// ---------------------------------------------------------------------------
template <int MODE>
__global__ void gemm16(const __bf16* __restrict__ X, const __bf16* __restrict__ W,
                       const float* __restrict__ bias, void* __restrict__ Yv,
                       int N, float oscale) {
  const int K = D_MODEL;
  int wave  = threadIdx.x >> 5;
  int lane  = threadIdx.x & 31;
  int row0  = blockIdx.x * 16;
  int col0  = (blockIdx.y * (blockDim.x >> 5) + wave) * 64;
  int m     = lane & 15;
  int hi    = lane >> 4;
  int khalf = hi * 8;

  v8f zero = {0.f, 0.f, 0.f, 0.f, 0.f, 0.f, 0.f, 0.f};
  v8f acc[4];
#pragma unroll
  for (int t = 0; t < 4; ++t) acc[t] = zero;

  const __bf16* xrow = X + (size_t)(row0 + m) * K;
  for (int kk = 0; kk < K; kk += 32) {
    v16bf a = load_frag_bf16(xrow + kk, khalf);
#pragma unroll
    for (int t = 0; t < 4; ++t) {
      int n = col0 + t * 16 + m;
      if (MODE == 2 && n >= N) n = N - 1;  // clamp OOB rows; stores guarded
      v16bf b = load_frag_bf16(W + (size_t)n * K + kk, khalf);
      acc[t] = wmma_bf16(a, b, acc[t]);
    }
  }

  // C/D layout: lane holds column n = lane&15, rows m = j + 8*(lane>=16)
#pragma unroll
  for (int t = 0; t < 4; ++t) {
#pragma unroll
    for (int j = 0; j < 8; ++j) {
      int mm = row0 + j + 8 * hi;
      int nn = col0 + t * 16 + m;
      float v = acc[t][j];
      if (MODE == 0) {
        ((__bf16*)Yv)[(size_t)mm * N + nn] = (__bf16)(v * oscale);
      } else if (MODE == 1) {
        int b_ = mm >> 11, s_ = mm & (SEQ - 1);
        int h_ = nn >> 5,  d_ = nn & 31;
        ((__bf16*)Yv)[(size_t)((b_ * NHEAD + h_) * DKH + d_) * SEQ + s_] = (__bf16)v;
      } else {
        if (nn < N) ((float*)Yv)[(size_t)mm * N + nn] = v + bias[nn];
      }
    }
  }
}

// ---------------------------------------------------------------------------
// One 32-key attention chunk. MASKED=false for chunks fully below the
// causal diagonal (no compares/selects). Lane-local softmax partial sums:
// only the row-max needs a cross-lane butterfly inside the loop; the
// denominator is reduced once at the epilogue.
// ---------------------------------------------------------------------------
template <bool MASKED>
__device__ __forceinline__ void attn_chunk(
    const __bf16* __restrict__ Km, const __bf16* __restrict__ Vt,
    int base_row, int bh, int kb, int qb, int m, int hi, int khalf,
    v16bf qf, __bf16 (*__restrict__ pb)[40],
    v8f& o0, v8f& o1, float* rmax, float* lsum) {
  const int S = SEQ, D = D_MODEL;
  v8f zero = {0.f, 0.f, 0.f, 0.f, 0.f, 0.f, 0.f, 0.f};

  const __bf16* krow = Km + (size_t)(base_row + kb + m) * D;
  v16bf kf0 = load_frag_bf16(krow, khalf);
  v16bf kf1 = load_frag_bf16(krow + (size_t)16 * D, khalf);
  // speculative prefetch of next key block (global_prefetch_b8)
  __builtin_prefetch(krow + (size_t)32 * D, 0, 3);

  v8f s0 = wmma_bf16(qf, kf0, zero);
  v8f s1 = wmma_bf16(qf, kf1, zero);

#pragma unroll
  for (int j = 0; j < 8; ++j) {
    float a0, a1;
    if (MASKED) {
      int qrow = qb + j + 8 * hi;
      a0 = (kb + m      <= qrow) ? s0[j] : -1e30f;   // causal mask
      a1 = (kb + 16 + m <= qrow) ? s1[j] : -1e30f;
    } else {
      a0 = s0[j];
      a1 = s1[j];
    }
    float cm = fmaxf(a0, a1);
#pragma unroll
    for (int off = 1; off < 16; off <<= 1)           // 16-lane half = one row
      cm = fmaxf(cm, __shfl_xor(cm, off, 32));
    float nm   = fmaxf(rmax[j], cm);
    float corr = __expf(rmax[j] - nm);               // row-uniform
    float p0   = __expf(a0 - nm);
    float p1   = __expf(a1 - nm);
    lsum[j] = lsum[j] * corr + p0 + p1;              // lane-local partial
    rmax[j] = nm;
    o0[j] *= corr;
    o1[j] *= corr;
    int rowm = j + 8 * hi;
    pb[rowm][m]      = (__bf16)p0;                   // C-layout -> LDS
    pb[rowm][16 + m] = (__bf16)p1;
  }

  // intra-wave LDS RAW fence (DS ops are in-order per wave; wait to be safe)
  asm volatile("s_wait_dscnt 0" ::: "memory");

  // P as A-fragment: two 128-bit DS reads per lane, no conversions
  v8bf plo = *(const v8bf*)&pb[m][khalf];
  v8bf phi = *(const v8bf*)&pb[m][16 + khalf];
  v16bf pf;
#pragma unroll
  for (int i = 0; i < 8; ++i) { pf[i] = plo[i]; pf[i + 8] = phi[i]; }

  const __bf16* vrow = Vt + (size_t)(bh * DKH + m) * S + kb;
  v16bf vf0 = load_frag_bf16(vrow, khalf);
  v16bf vf1 = load_frag_bf16(vrow + (size_t)16 * S, khalf);
  o0 = wmma_bf16(pf, vf0, o0);
  o1 = wmma_bf16(pf, vf1, o1);
}

// ---------------------------------------------------------------------------
// Flash-style causal attention. One wave per 16-query block of one (b,h).
// Q,K: [B*S, 256] bf16 (head h at column h*32; softmax scale pre-folded
// into Q). Vt: [(b*H+h)*32 + d][S] bf16.
// ---------------------------------------------------------------------------
__global__ void attn_kernel(const __bf16* __restrict__ Q,
                            const __bf16* __restrict__ Km,
                            const __bf16* __restrict__ Vt,
                            __bf16* __restrict__ O) {
  const int S = SEQ, D = D_MODEL;
  int wid  = (blockIdx.x * blockDim.x + threadIdx.x) >> 5;
  int lane = threadIdx.x & 31;
  int qblk = wid & 127;        // S/16 = 128 query blocks
  int bh   = wid >> 7;         // 0..31
  int b    = bh >> 3;
  int h    = bh & 7;
  int qb   = qblk * 16;

  __shared__ alignas(16) __bf16 pbuf[4][16][40];   // per-wave P staging (+pad)
  __bf16 (*pb)[40] = pbuf[threadIdx.x >> 5];

  int m = lane & 15, hi = lane >> 4, khalf = hi * 8;
  int base_row = b * S;        // head column folded into row pointer below

  v16bf qf = load_frag_bf16(Q + (size_t)(base_row + qb + m) * D + h * DKH, khalf);

  v8f zero = {0.f, 0.f, 0.f, 0.f, 0.f, 0.f, 0.f, 0.f};
  v8f o0 = zero, o1 = zero;
  float rmax[8], lsum[8];
#pragma unroll
  for (int j = 0; j < 8; ++j) { rmax[j] = -1e30f; lsum[j] = 0.f; }

  const __bf16* Kh = Km + h * DKH;  // fold head offset once

  // chunks fully below the diagonal (kb+31 <= qb): no masking needed
  int kb_full_end = (qb >= 31) ? ((((qb - 31) >> 5) << 5) + 32) : 0;
  for (int kb = 0; kb < kb_full_end; kb += 32)       // wave-uniform bound
    attn_chunk<false>(Kh, Vt, base_row, bh, kb, qb, m, hi, khalf,
                      qf, pb, o0, o1, rmax, lsum);
  // diagonal-straddling chunks: apply causal mask
  for (int kb = kb_full_end; kb < qb + 16; kb += 32) // wave-uniform bound
    attn_chunk<true>(Kh, Vt, base_row, bh, kb, qb, m, hi, khalf,
                     qf, pb, o0, o1, rmax, lsum);

  // epilogue: reduce lane-local softmax sums once, normalize, store bf16
#pragma unroll
  for (int j = 0; j < 8; ++j) {
    float ls = lsum[j];
#pragma unroll
    for (int off = 1; off < 16; off <<= 1)
      ls += __shfl_xor(ls, off, 32);
    float inv = 1.f / ls;
    size_t mm = (size_t)(base_row + qb + j + 8 * hi) * D + h * DKH;
    O[mm + m]      = (__bf16)(o0[j] * inv);
    O[mm + 16 + m] = (__bf16)(o1[j] * inv);
  }
}

// ---------------------------------------------------------------------------
// LayerNorm over last dim (256), f32 math, bf16 in/out. One wave per row.
// ---------------------------------------------------------------------------
__global__ void ln_kernel(const __bf16* __restrict__ Hm,
                          const float* __restrict__ gamma,
                          const float* __restrict__ beta,
                          __bf16* __restrict__ Yn) {
  int row  = (blockIdx.x * blockDim.x + threadIdx.x) >> 5;
  int lane = threadIdx.x & 31;
  const __bf16* r = Hm + (size_t)row * D_MODEL;
  float v[8], s = 0.f;
#pragma unroll
  for (int i = 0; i < 8; ++i) { v[i] = (float)r[i * 32 + lane]; s += v[i]; }
#pragma unroll
  for (int off = 1; off < 32; off <<= 1) s += __shfl_xor(s, off, 32);
  float mean = s * (1.0f / D_MODEL);
  float var = 0.f;
#pragma unroll
  for (int i = 0; i < 8; ++i) { float d = v[i] - mean; var += d * d; }
#pragma unroll
  for (int off = 1; off < 32; off <<= 1) var += __shfl_xor(var, off, 32);
  var *= (1.0f / D_MODEL);
  float inv = rsqrtf(var + 1e-5f);
#pragma unroll
  for (int i = 0; i < 8; ++i) {
    int c = i * 32 + lane;
    Yn[(size_t)row * D_MODEL + c] =
        (__bf16)((v[i] - mean) * inv * gamma[c] + beta[c]);
  }
}

// ---------------------------------------------------------------------------
// inputs: x, embed, Wq, Wk, Wv, Wo, gamma, beta, Whead, bhead
// output: [B, S, VOCAB] f32
// ---------------------------------------------------------------------------
extern "C" void kernel_launch(void* const* d_in, const int* in_sizes, int n_in,
                              void* d_out, int out_size, void* d_ws, size_t ws_size,
                              hipStream_t stream) {
  (void)in_sizes; (void)n_in; (void)out_size; (void)ws_size;

  const int*   x     = (const int*)  d_in[0];
  const float* emb   = (const float*)d_in[1];
  const float* Wq    = (const float*)d_in[2];
  const float* Wk    = (const float*)d_in[3];
  const float* Wv    = (const float*)d_in[4];
  const float* Wo    = (const float*)d_in[5];
  const float* gamma = (const float*)d_in[6];
  const float* beta  = (const float*)d_in[7];
  const float* Whead = (const float*)d_in[8];
  const float* bhead = (const float*)d_in[9];
  float*       out   = (float*)d_out;

  const size_t actE = (size_t)ROWS * D_MODEL;             // 2M elems
  const size_t wE   = (size_t)NLAYER * D_MODEL * D_MODEL; // 262144 elems
  const size_t whE  = (size_t)VOCABN * D_MODEL;           // 256000 elems

  __bf16* ws   = (__bf16*)d_ws;
  __bf16* hb   = ws;                 // hidden state (reused every layer)
  __bf16* qb   = hb + actE;
  __bf16* kb   = qb + actE;
  __bf16* vt   = kb + actE;          // per-head transposed V
  __bf16* ao   = vt + actE;          // attention output
  __bf16* bWq  = ao + actE;
  __bf16* bWk  = bWq + wE;
  __bf16* bWv  = bWk + wE;
  __bf16* bWo  = bWv + wE;
  __bf16* bWh  = bWo + wE;

  // one-time weight conversion to bf16 (L2-resident afterwards)
  cvt_bf16_kernel<<<(int)((wE + 255) / 256), 256, 0, stream>>>(Wq, bWq, (int)wE);
  cvt_bf16_kernel<<<(int)((wE + 255) / 256), 256, 0, stream>>>(Wk, bWk, (int)wE);
  cvt_bf16_kernel<<<(int)((wE + 255) / 256), 256, 0, stream>>>(Wv, bWv, (int)wE);
  cvt_bf16_kernel<<<(int)((wE + 255) / 256), 256, 0, stream>>>(Wo, bWo, (int)wE);
  cvt_bf16_kernel<<<(int)((whE + 255) / 256), 256, 0, stream>>>(Whead, bWh, (int)whE);

  embed_kernel<<<ROWS, D_MODEL, 0, stream>>>(x, emb, hb);

  const float qscale = 0.17677669529663687f;  // 1/sqrt(dk)
  const dim3 gProj(ROWS / 16, 1);             // 4 waves x 64 cols = full 256
  for (int L = 0; L < NLAYER; ++L) {
    const __bf16* wq = bWq + (size_t)L * D_MODEL * D_MODEL;
    const __bf16* wk = bWk + (size_t)L * D_MODEL * D_MODEL;
    const __bf16* wv = bWv + (size_t)L * D_MODEL * D_MODEL;
    const __bf16* wo = bWo + (size_t)L * D_MODEL * D_MODEL;

    gemm16<0><<<gProj, 128, 0, stream>>>(hb, wq, nullptr, qb, D_MODEL, qscale);
    gemm16<0><<<gProj, 128, 0, stream>>>(hb, wk, nullptr, kb, D_MODEL, 1.0f);
    gemm16<1><<<gProj, 128, 0, stream>>>(hb, wv, nullptr, vt, D_MODEL, 1.0f);

    // 4096 query-block waves, 4 waves/block
    attn_kernel<<<1024, 128, 0, stream>>>(qb, kb, vt, ao);

    gemm16<0><<<gProj, 128, 0, stream>>>(ao, wo, nullptr, hb, D_MODEL, 1.0f);
  }

  // LayerNorm into qb (free now), then vocab head
  ln_kernel<<<ROWS / 8, 256, 0, stream>>>(hb, gamma, beta, qb);

  // 16 waves * 64 cols = 1024 >= 1000; all waves have in-range work
  gemm16<2><<<dim3(ROWS / 16, 4), 128, 0, stream>>>(qb, bWh, bhead, out,
                                                    VOCABN, 1.0f);
}
